// SimpleVAE_12232066859406
// MI455X (gfx1250) — matrix operand order repaired and checked by reading, not compiled, not run on hardware
//
#include <hip/hip_runtime.h>
#include <hip/hip_bf16.h>
#include <stdint.h>

typedef unsigned short u16_t;
typedef __attribute__((ext_vector_type(16))) __bf16 v16bf;
typedef __attribute__((ext_vector_type(8)))  float  v8f;

#define BATCH 128
#define SIN   256
#define SOUT  128
#define DIN   128
#define DOUT  128
#define HH    512
#define LATD  64

__device__ __forceinline__ u16_t f2bf(float f) {
  uint32_t u = __float_as_uint(f);
  u += 0x7FFFu + ((u >> 16) & 1u);          // round-to-nearest-even truncation
  return (u16_t)(u >> 16);
}
__device__ __forceinline__ float sigmoidf_(float x) { return 1.0f / (1.0f + __expf(-x)); }

union BfFrag { uint4 q[2]; v16bf v; };

// A-matrix 16x32 bf16 fragment (lanes 0-15: M=0-15 / K=base..+15; lanes 16-31 same M, K+16..+31)
__device__ __forceinline__ v16bf load_afrag(const u16_t* __restrict__ A, int ldA,
                                            int m0, int kc, int lane) {
  int m = m0 + (lane & 15);
  int koff = (lane >> 4) << 4;
  const u16_t* p = A + (size_t)m * ldA + kc + koff;
  BfFrag f;
  f.q[0] = *(const uint4*)(p);       // K = kc+koff .. +7
  f.q[1] = *(const uint4*)(p + 8);   // K = kc+koff+8 .. +15
  return f.v;
}

// B-matrix 32x16 bf16 fragment from row-major W[N][K] (B = W^T): lane n = nbase+(lane&15),
// 16 contiguous halves of K starting at kc + (lane>=16 ? 16 : 0)
__device__ __forceinline__ v16bf load_bfrag(const u16_t* __restrict__ W, int ldW,
                                            int nbase, int kc, int lane) {
  int n = nbase + (lane & 15);
  const u16_t* p = W + (size_t)n * ldW + kc + ((lane >> 4) << 4);
  BfFrag f;
  f.q[0] = *(const uint4*)(p);
  f.q[1] = *(const uint4*)(p + 8);
  return f.v;
}

#define WMMA_BF16(acc, a, b) \
  (acc) = __builtin_amdgcn_wmma_f32_16x16x32_bf16(false, (a), false, (b), (short)0, (acc), false, false)

// ---------------------------------------------------------------------------
// Fused GRU cell: h_new = (1-z)*tanh(gi_n + r*gh_n) + z*h_prev
//   GI = A1 @ Wih^T (+gx precomputed constant part), GH = Hprev @ Whh^T
// One wave owns a 16x32 tile of h_new -> 128 waves = 16 blocks x 256 threads.
// ---------------------------------------------------------------------------
__global__ __launch_bounds__(256) void gru_cell_wmma(
    const u16_t* __restrict__ A1, int ldA1, int K1,
    const u16_t* __restrict__ Hbf,
    const u16_t* __restrict__ Wih, int ldWih,
    const u16_t* __restrict__ Whh,
    const float* __restrict__ bih, const float* __restrict__ bhh,
    const float* __restrict__ gx,            // [128 x 1536] constant GI addend or null
    const float* __restrict__ Hf,
    float* __restrict__ Hnf, u16_t* __restrict__ Hnbf)
{
  int wv   = blockIdx.x * (blockDim.x >> 5) + (threadIdx.x >> 5);
  int lane = threadIdx.x & 31;
  int m0 = (wv & 7) << 4;          // row tile (batch)
  int j0 = (wv >> 3) << 5;         // 32 h-columns

  v8f aR0 = {}, aR1 = {}, aZ0 = {}, aZ1 = {};
  v8f aI0 = {}, aI1 = {}, aN0 = {}, aN1 = {};

  // GI: A1 x Wih^T  (r & z accumulate jointly with GH later; i kept separate)
  for (int kc = 0; kc < K1; kc += 32) {
    v16bf a  = load_afrag(A1, ldA1, m0, kc, lane);
    v16bf r0 = load_bfrag(Wih, ldWih, j0,            kc, lane);
    v16bf r1 = load_bfrag(Wih, ldWih, j0 + 16,       kc, lane);
    v16bf z0 = load_bfrag(Wih, ldWih, HH + j0,       kc, lane);
    v16bf z1 = load_bfrag(Wih, ldWih, HH + j0 + 16,  kc, lane);
    v16bf i0 = load_bfrag(Wih, ldWih, 2*HH + j0,     kc, lane);
    v16bf i1 = load_bfrag(Wih, ldWih, 2*HH + j0 + 16,kc, lane);
    WMMA_BF16(aR0, a, r0); WMMA_BF16(aR1, a, r1);
    WMMA_BF16(aZ0, a, z0); WMMA_BF16(aZ1, a, z1);
    WMMA_BF16(aI0, a, i0); WMMA_BF16(aI1, a, i1);
  }
  // GH: Hprev x Whh^T
  for (int kc = 0; kc < HH; kc += 32) {
    v16bf a  = load_afrag(Hbf, HH, m0, kc, lane);
    v16bf r0 = load_bfrag(Whh, HH, j0,            kc, lane);
    v16bf r1 = load_bfrag(Whh, HH, j0 + 16,       kc, lane);
    v16bf z0 = load_bfrag(Whh, HH, HH + j0,       kc, lane);
    v16bf z1 = load_bfrag(Whh, HH, HH + j0 + 16,  kc, lane);
    v16bf n0 = load_bfrag(Whh, HH, 2*HH + j0,     kc, lane);
    v16bf n1 = load_bfrag(Whh, HH, 2*HH + j0 + 16,kc, lane);
    WMMA_BF16(aR0, a, r0); WMMA_BF16(aR1, a, r1);
    WMMA_BF16(aZ0, a, z0); WMMA_BF16(aZ1, a, z1);
    WMMA_BF16(aN0, a, n0); WMMA_BF16(aN1, a, n1);
  }

  int ms = (lane >> 4) << 3;
  int nn = lane & 15;
  for (int s = 0; s < 2; ++s) {
    v8f vR = s ? aR1 : aR0, vZ = s ? aZ1 : aZ0;
    v8f vI = s ? aI1 : aI0, vN = s ? aN1 : aN0;
    int j = j0 + s * 16 + nn;
    float bR = bih[j] + bhh[j];
    float bZ = bih[HH + j] + bhh[HH + j];
    float bI = bih[2*HH + j];
    float bH = bhh[2*HH + j];
    for (int e = 0; e < 8; ++e) {
      int m = m0 + ms + e;
      float gr = vR[e] + bR, gz = vZ[e] + bZ, gi = vI[e] + bI, gh = vN[e] + bH;
      if (gx) {
        const float* g = gx + (size_t)m * (3 * HH);
        gr += g[j]; gz += g[HH + j]; gi += g[2*HH + j];
      }
      float r = sigmoidf_(gr);
      float z = sigmoidf_(gz);
      float n = tanhf(gi + r * gh);
      float hp = Hf[(size_t)m * HH + j];
      float h  = (1.0f - z) * n + z * hp;
      Hnf [(size_t)m * HH + j] = h;
      Hnbf[(size_t)m * HH + j] = f2bf(h);
    }
  }
}

// ---------------------------------------------------------------------------
// Generic WMMA GEMM: C = act( A1@W1^T [+ A2@W2^T] + bias + addend )
// Wave owns 16x32 tile; total waves = 8 * (N/32); launch N/32 blocks of 256.
// ---------------------------------------------------------------------------
__global__ __launch_bounds__(256) void gemm_wmma(
    const u16_t* __restrict__ A1, int ldA1, int K1,
    const u16_t* __restrict__ A2, int ldA2, int K2,
    const u16_t* __restrict__ W1, int ldW1,
    const u16_t* __restrict__ W2, int ldW2,
    const float* __restrict__ bias,
    const float* __restrict__ addend, int ldAdd,
    int relu,
    float* __restrict__ Cf, int ldCf,
    u16_t* __restrict__ Cb, int ldCb,
    int N)
{
  int wv   = blockIdx.x * (blockDim.x >> 5) + (threadIdx.x >> 5);
  int lane = threadIdx.x & 31;
  int mg = wv & 7;
  int ng = wv >> 3;
  if (ng >= (N >> 5)) return;
  int m0 = mg << 4, n0 = ng << 5;

  v8f acc0 = {}, acc1 = {};
  for (int kc = 0; kc < K1; kc += 32) {
    v16bf a  = load_afrag(A1, ldA1, m0, kc, lane);
    v16bf b0 = load_bfrag(W1, ldW1, n0,      kc, lane);
    v16bf b1 = load_bfrag(W1, ldW1, n0 + 16, kc, lane);
    WMMA_BF16(acc0, a, b0);
    WMMA_BF16(acc1, a, b1);
  }
  if (A2) {
    for (int kc = 0; kc < K2; kc += 32) {
      v16bf a  = load_afrag(A2, ldA2, m0, kc, lane);
      v16bf b0 = load_bfrag(W2, ldW2, n0,      kc, lane);
      v16bf b1 = load_bfrag(W2, ldW2, n0 + 16, kc, lane);
      WMMA_BF16(acc0, a, b0);
      WMMA_BF16(acc1, a, b1);
    }
  }

  int ms = (lane >> 4) << 3;
  int nn = lane & 15;
  for (int s = 0; s < 2; ++s) {
    v8f acc = s ? acc1 : acc0;
    int n = n0 + s * 16 + nn;
    float bs = bias ? bias[n] : 0.0f;
    for (int e = 0; e < 8; ++e) {
      int m = m0 + ms + e;
      float c = acc[e] + bs;
      if (addend) c += addend[(size_t)m * ldAdd + n];
      if (relu)   c = fmaxf(c, 0.0f);
      if (Cf) Cf[(size_t)m * ldCf + n] = c;
      if (Cb) Cb[(size_t)m * ldCb + n] = f2bf(c);
    }
  }
}

// ---------------------------------------------------------------------------
// Latent path (f32 scalar; tiny): mu, logvar, z = mu + eps*exp(0.5*lv),
// hid = z @ di_W^T + di_b  -> decoder initial states (f32 + bf16)
// ---------------------------------------------------------------------------
__global__ __launch_bounds__(256) void latent_kernel(
    const float* __restrict__ lh,
    const float* __restrict__ muW, const float* __restrict__ mub,
    const float* __restrict__ lvW, const float* __restrict__ lvb,
    const float* __restrict__ eps,
    const float* __restrict__ diW, const float* __restrict__ dib,
    float* __restrict__ mu_out, float* __restrict__ lv_out,
    float* __restrict__ h0f, u16_t* __restrict__ h0b,
    float* __restrict__ h1f, u16_t* __restrict__ h1b)
{
  __shared__ float s_lh[HH];
  __shared__ float s_z[LATD];
  int b = blockIdx.x;
  for (int i = threadIdx.x; i < HH; i += blockDim.x) s_lh[i] = lh[(size_t)b * HH + i];
  __syncthreads();
  if (threadIdx.x < LATD) {
    int j = threadIdx.x;
    const float* wm = muW + (size_t)j * HH;
    const float* wl = lvW + (size_t)j * HH;
    float am = 0.f, al = 0.f;
    for (int k = 0; k < HH; ++k) { am += s_lh[k] * wm[k]; al += s_lh[k] * wl[k]; }
    float mu = am + mub[j], lv = al + lvb[j];
    mu_out[(size_t)b * LATD + j] = mu;
    lv_out[(size_t)b * LATD + j] = lv;
    s_z[j] = mu + eps[(size_t)b * LATD + j] * __expf(0.5f * lv);
  }
  __syncthreads();
  for (int g = threadIdx.x; g < 2 * HH; g += blockDim.x) {
    const float* w = diW + (size_t)g * LATD;
    float acc = dib[g];
    for (int k = 0; k < LATD; ++k) acc += s_z[k] * w[k];
    if (g < HH) { h0f[(size_t)b * HH + g] = acc;      h0b[(size_t)b * HH + g] = f2bf(acc); }
    else { int g2 = g - HH; h1f[(size_t)b * HH + g2] = acc; h1b[(size_t)b * HH + g2] = f2bf(acc); }
  }
}

__global__ void cvt_f32_bf16(const float* __restrict__ s, u16_t* __restrict__ d, size_t n) {
  size_t i = (size_t)blockIdx.x * blockDim.x + threadIdx.x;
  size_t stride = (size_t)gridDim.x * blockDim.x;
  for (; i < n; i += stride) d[i] = f2bf(s[i]);
}

// ---------------------------------------------------------------------------
extern "C" void kernel_launch(void* const* d_in, const int* in_sizes, int n_in,
                              void* d_out, int out_size, void* d_ws, size_t ws_size,
                              hipStream_t stream) {
  (void)in_sizes; (void)n_in; (void)out_size; (void)ws_size;
  const float* x        = (const float*)d_in[0];
  const float* eps      = (const float*)d_in[2];
  const float* eWih0    = (const float*)d_in[3];
  const float* eWhh0    = (const float*)d_in[4];
  const float* eBih0    = (const float*)d_in[5];
  const float* eBhh0    = (const float*)d_in[6];
  const float* eWih1    = (const float*)d_in[7];
  const float* eWhh1    = (const float*)d_in[8];
  const float* eBih1    = (const float*)d_in[9];
  const float* eBhh1    = (const float*)d_in[10];
  const float* muW      = (const float*)d_in[11];
  const float* mub      = (const float*)d_in[12];
  const float* lvW      = (const float*)d_in[13];
  const float* lvb      = (const float*)d_in[14];
  const float* diW      = (const float*)d_in[15];
  const float* dib      = (const float*)d_in[16];
  // d_in[17..19] (attn_W, attn_b, v_w) unused: attention over identical rows
  // reduces exactly to context == last_hidden (softmax of a constant is uniform).
  const float* dWih0    = (const float*)d_in[20];
  const float* dWhh0    = (const float*)d_in[21];
  const float* dBih0    = (const float*)d_in[22];
  const float* dBhh0    = (const float*)d_in[23];
  const float* dWih1    = (const float*)d_in[24];
  const float* dWhh1    = (const float*)d_in[25];
  const float* dBih1    = (const float*)d_in[26];
  const float* dBhh1    = (const float*)d_in[27];
  const float* o1W      = (const float*)d_in[28];
  const float* o1b      = (const float*)d_in[29];
  const float* o2W      = (const float*)d_in[30];
  const float* o2b      = (const float*)d_in[31];

  float* out_recon = (float*)d_out;                              // [128,128,128]
  float* out_mu    = out_recon + (size_t)BATCH * SOUT * DOUT;    // [128,64]
  float* out_lv    = out_mu + (size_t)BATCH * LATD;              // [128,64]

  // -- carve workspace ------------------------------------------------------
  uint8_t* wsp = (uint8_t*)d_ws;
  auto alloc = [&](size_t bytes) -> void* {
    void* r = (void*)wsp;
    wsp += (bytes + 255) & ~(size_t)255;
    return r;
  };
  u16_t* xbf   = (u16_t*)alloc((size_t)BATCH * SIN * DIN * 2);
  u16_t* w_ih0 = (u16_t*)alloc((size_t)3*HH * DIN * 2);
  u16_t* w_hh0 = (u16_t*)alloc((size_t)3*HH * HH * 2);
  u16_t* w_ih1 = (u16_t*)alloc((size_t)3*HH * HH * 2);
  u16_t* w_hh1 = (u16_t*)alloc((size_t)3*HH * HH * 2);
  u16_t* dw_ih0= (u16_t*)alloc((size_t)3*HH * (DOUT + HH) * 2); // ld 640
  u16_t* dw_hh0= (u16_t*)alloc((size_t)3*HH * HH * 2);
  u16_t* dw_ih1= (u16_t*)alloc((size_t)3*HH * HH * 2);
  u16_t* dw_hh1= (u16_t*)alloc((size_t)3*HH * HH * 2);
  u16_t* o1w   = (u16_t*)alloc((size_t)HH * (2*HH + DOUT) * 2); // ld 1152
  u16_t* o2w   = (u16_t*)alloc((size_t)DOUT * HH * 2);

  float* h0f[2];  float* h1f[2];  u16_t* h0b[2];  u16_t* h1b[2];
  float* g0f[2];  float* g1f[2];  u16_t* g0b[2];  u16_t* g1b[2];
  for (int i = 0; i < 2; ++i) {
    h0f[i] = (float*)alloc((size_t)BATCH * HH * 4);
    h1f[i] = (float*)alloc((size_t)BATCH * HH * 4);
    h0b[i] = (u16_t*)alloc((size_t)BATCH * HH * 2);
    h1b[i] = (u16_t*)alloc((size_t)BATCH * HH * 2);
    g0f[i] = (float*)alloc((size_t)BATCH * HH * 4);
    g1f[i] = (float*)alloc((size_t)BATCH * HH * 4);
    g0b[i] = (u16_t*)alloc((size_t)BATCH * HH * 2);
    g1b[i] = (u16_t*)alloc((size_t)BATCH * HH * 2);
  }
  u16_t* decin = (u16_t*)alloc((size_t)BATCH * DOUT * 2);
  float* gctx  = (float*)alloc((size_t)BATCH * 3 * HH * 4);   // lh @ dec_Wih0[:,128:640]^T
  float* octx  = (float*)alloc((size_t)BATCH * HH * 4);       // lh @ out1_W[:,512:1024]^T
  u16_t* hmid  = (u16_t*)alloc((size_t)BATCH * HH * 2);

  // -- one-time conversions / zero init ------------------------------------
  auto cvt = [&](const float* s, u16_t* d, size_t n) {
    cvt_f32_bf16<<<512, 256, 0, stream>>>(s, d, n);
  };
  cvt(x,     xbf,   (size_t)BATCH * SIN * DIN);
  cvt(eWih0, w_ih0, (size_t)3*HH * DIN);
  cvt(eWhh0, w_hh0, (size_t)3*HH * HH);
  cvt(eWih1, w_ih1, (size_t)3*HH * HH);
  cvt(eWhh1, w_hh1, (size_t)3*HH * HH);
  cvt(dWih0, dw_ih0,(size_t)3*HH * (DOUT + HH));
  cvt(dWhh0, dw_hh0,(size_t)3*HH * HH);
  cvt(dWih1, dw_ih1,(size_t)3*HH * HH);
  cvt(dWhh1, dw_hh1,(size_t)3*HH * HH);
  cvt(o1W,   o1w,   (size_t)HH * (2*HH + DOUT));
  cvt(o2W,   o2w,   (size_t)DOUT * HH);

  hipMemsetAsync(h0f[0], 0, (size_t)BATCH * HH * 4, stream);
  hipMemsetAsync(h1f[0], 0, (size_t)BATCH * HH * 4, stream);
  hipMemsetAsync(h0b[0], 0, (size_t)BATCH * HH * 2, stream);
  hipMemsetAsync(h1b[0], 0, (size_t)BATCH * HH * 2, stream);
  hipMemsetAsync(decin,  0, (size_t)BATCH * DOUT * 2, stream);

  // -- encoder scan ---------------------------------------------------------
  int cur = 0;
  for (int t = 0; t < SIN; ++t) {
    int nxt = cur ^ 1;
    gru_cell_wmma<<<16, 256, 0, stream>>>(
        xbf + (size_t)t * DIN, SIN * DIN, DIN,
        h0b[cur], w_ih0, DIN, w_hh0, eBih0, eBhh0, nullptr,
        h0f[cur], h0f[nxt], h0b[nxt]);
    gru_cell_wmma<<<16, 256, 0, stream>>>(
        h0b[nxt], HH, HH,
        h1b[cur], w_ih1, HH, w_hh1, eBih1, eBhh1, nullptr,
        h1f[cur], h1f[nxt], h1b[nxt]);
    cur = nxt;
  }
  const float* lhf = h1f[cur];
  const u16_t* lhb = h1b[cur];

  // -- latent: mu/logvar/z, decoder initial states --------------------------
  latent_kernel<<<BATCH, 256, 0, stream>>>(
      lhf, muW, mub, lvW, lvb, eps, diW, dib,
      out_mu, out_lv, g0f[0], g0b[0], g1f[0], g1b[0]);

  // -- hoist constant-context GEMM contributions ----------------------------
  // gctx = lh @ dec_Wih0[:,128:640]^T     (context part of decoder cell0 input)
  gemm_wmma<<<48, 256, 0, stream>>>(
      lhb, HH, HH, nullptr, 0, 0,
      dw_ih0 + DOUT, DOUT + HH, nullptr, 0,
      nullptr, nullptr, 0, 0,
      gctx, 3 * HH, nullptr, 0, 3 * HH);
  // octx = lh @ out1_W[:,512:1024]^T      (context part of output MLP)
  gemm_wmma<<<16, 256, 0, stream>>>(
      lhb, HH, HH, nullptr, 0, 0,
      o1w + HH, 2*HH + DOUT, nullptr, 0,
      nullptr, nullptr, 0, 0,
      octx, HH, nullptr, 0, HH);

  // -- decoder scan ---------------------------------------------------------
  int dc = 0;
  for (int t = 0; t < SOUT; ++t) {
    int dn = dc ^ 1;
    // cell0: gin = [dec_in, context];  dec_in part live, context part = gctx
    gru_cell_wmma<<<16, 256, 0, stream>>>(
        decin, DOUT, DOUT,
        g0b[dc], dw_ih0, DOUT + HH, dw_hh0, dBih0, dBhh0, gctx,
        g0f[dc], g0f[dn], g0b[dn]);
    // cell1
    gru_cell_wmma<<<16, 256, 0, stream>>>(
        g0b[dn], HH, HH,
        g1b[dc], dw_ih1, HH, dw_hh1, dBih1, dBhh1, nullptr,
        g1f[dc], g1f[dn], g1b[dn]);
    // out1: hmid = relu(h1@Wo_h^T + dec_in@Wo_x^T + octx + b)
    gemm_wmma<<<16, 256, 0, stream>>>(
        g1b[dn], HH, HH, decin, DOUT, DOUT,
        o1w, 2*HH + DOUT, o1w + 2*HH, 2*HH + DOUT,
        o1b, octx, HH, 1,
        nullptr, 0, hmid, HH, HH);
    // out2: out_vec = relu(hmid@out2_W^T + b); write recon slice + next dec_in
    gemm_wmma<<<4, 256, 0, stream>>>(
        hmid, HH, HH, nullptr, 0, 0,
        o2w, HH, nullptr, 0,
        o2b, nullptr, 0, 1,
        out_recon + (size_t)t * DOUT, SOUT * DOUT, decin, DOUT, DOUT);
    dc = dn;
  }
}
